// OTransformer_47614007443737
// MI455X (gfx1250) — compile-verified, matrix operand order checked by reading, not run on hardware
//
#include <hip/hip_runtime.h>
#include <hip/hip_bf16.h>

// Model dims (match reference)
#define B_   4
#define S_   1024
#define D_   512
#define H_   8
#define L_   4
#define DFF_ 2048
#define V_   32000
#define DK_  64
#define KSEL_ 307
#define M_   (B_*S_)   // 4096 rows of activations

typedef __attribute__((ext_vector_type(16))) __bf16 v16bf;
typedef __attribute__((ext_vector_type(8)))  float  v8f;
typedef __attribute__((ext_vector_type(4)))  unsigned int v4u;
typedef __attribute__((ext_vector_type(2)))  unsigned int v2u;

union FragU { v4u q[2]; v16bf v; };

enum { FLAG_TB = 1, FLAG_BIAS = 2, FLAG_GELU = 4, FLAG_RES = 8 };

__device__ __forceinline__ unsigned int bfround(unsigned int u) {
    return u + 0x7FFFu + ((u >> 16) & 1u);       // RNE round to bf16 in [31:16]
}
// pack two f32 -> bf16x2 with one v_perm_b32 (bytes: lo=ra[3:2], hi=rb[3:2])
__device__ __forceinline__ unsigned int pack2bf(float lo, float hi) {
    return __builtin_amdgcn_perm(bfround(__float_as_uint(hi)),
                                 bfround(__float_as_uint(lo)), 0x07060302u);
}
__device__ __forceinline__ void pack_store(unsigned short* dst, const float4 v) {
    v2u pk;
    pk.x = pack2bf(v.x, v.y);
    pk.y = pack2bf(v.z, v.w);
    *(v2u*)dst = pk;
}
__device__ __forceinline__ float gelu_exact(float x) {
    return 0.5f * x * (1.0f + erff(x * 0.70710678118654752f));
}

// ---------------------------------------------------------------------------
// WMMA GEMM, 128x128 block tile, BK=32, double-buffered LDS pipeline.
//   A: [M,Kd] f32, row stride lda
//   B: normal  -> [Kd,N] row stride ldb;  FLAG_TB -> [N,Kd] row stride ldb
// 8 wave32 waves; wave (wm,wn): rows wm*32..+32, cols wn*64..+64
//   -> 2x4 grid of 16x16 WMMA tiles = 8 v_wmma per wave per K-step.
// All M are multiples of 128. N=64 (attn@V) handled via clamps/guards;
// every TB-path N is a multiple of 128 (no guard there).
// ---------------------------------------------------------------------------
__global__ __launch_bounds__(256) void wmma_gemm_kernel(
    const float* __restrict__ A, const float* __restrict__ B,
    float* __restrict__ C, const float* __restrict__ bias,
    const float* __restrict__ R,
    int Nn, int Kd, int lda, int ldb, int ldc,
    long long sAz, long long sBz, long long sCz, long long sRz,
    float alpha, int flags)
{
    __shared__ __align__(16) unsigned short As[2][128 * 32];  // [row][k] bf16
    __shared__ __align__(16) unsigned short Bs[2][128 * 32];  // [n][k]   bf16

    const int tid = threadIdx.x;
    const int z   = blockIdx.z;
    A += (long long)z * sAz;
    B += (long long)z * sBz;
    C += (long long)z * sCz;
    if (flags & FLAG_RES) R += (long long)z * sRz;

    const int n0 = blockIdx.x * 128;
    const int m0 = blockIdx.y * 128;
    const int w = tid >> 5, lane = tid & 31;
    const int wm = w >> 1, wn = w & 1;          // 4x2 wave grid
    const int lm = lane & 15, hi = lane >> 4;   // ISA 16-bit frag lane split

    // Staging maps. A / B-transposed: 128 rows x 32 k, thread -> (row, 4 k)
    const int arow = tid >> 3;             // 0..31 (4 passes of 32 rows)
    const int acg  = (tid & 7) * 4;        // k group
    // B normal: thread -> fixed col n (4 passes of 32 cols), 4 consecutive k
    const int bn  = tid & 31;              // col within 32-group
    const int bkb = (tid >> 5) * 4;        // k base: 0,4,..,28

    v8f acc[2][4];
    #pragma unroll
    for (int i = 0; i < 2; ++i)
        #pragma unroll
        for (int j = 0; j < 4; ++j)
            acc[i][j] = v8f{0.f,0.f,0.f,0.f,0.f,0.f,0.f,0.f};

    auto loadA_g = [&](int k0, float4* r) {
        #pragma unroll
        for (int p = 0; p < 4; ++p)
            r[p] = *(const float4*)(A + (long long)(m0 + arow + p * 32) * lda + k0 + acg);
    };
    auto storeA_l = [&](unsigned short* buf, const float4* r) {
        #pragma unroll
        for (int p = 0; p < 4; ++p)
            pack_store(&buf[(arow + p * 32) * 32 + acg], r[p]);
    };
    auto loadB_g = [&](int k0, float4* r) {
        if (flags & FLAG_TB) {
            #pragma unroll
            for (int p = 0; p < 4; ++p)
                r[p] = *(const float4*)(B + (long long)(n0 + arow + p * 32) * ldb + k0 + acg);
        } else {
            #pragma unroll
            for (int q = 0; q < 4; ++q) {
                int col = bn + q * 32;                         // clamp for N=64 case
                if (col > Nn - n0 - 1) col = Nn - n0 - 1;
                const float* bp = B + n0 + col + (long long)(k0 + bkb) * ldb;
                r[q].x = bp[0];
                r[q].y = bp[(long long)ldb];
                r[q].z = bp[(long long)2 * ldb];
                r[q].w = bp[(long long)3 * ldb];
            }
        }
    };
    auto storeB_l = [&](unsigned short* buf, const float4* r) {
        if (flags & FLAG_TB) {
            #pragma unroll
            for (int p = 0; p < 4; ++p)
                pack_store(&buf[(arow + p * 32) * 32 + acg], r[p]);
        } else {
            #pragma unroll
            for (int q = 0; q < 4; ++q)
                pack_store(&buf[(bn + q * 32) * 32 + bkb], r[q]);
        }
    };

    // Prologue: stage k0 = 0 into buffer 0
    {
        float4 ra[4], rb[4];
        loadA_g(0, ra); loadB_g(0, rb);
        storeA_l(As[0], ra); storeB_l(Bs[0], rb);
    }
    __syncthreads();

    int cur = 0;
    for (int k0 = 0; k0 < Kd; k0 += 32) {
        const bool hasNext = (k0 + 32 < Kd);
        float4 ra[4], rb[4];
        if (hasNext) {                       // issue next tile's global loads
            loadA_g(k0 + 32, ra);
            loadB_g(k0 + 32, rb);
            __builtin_prefetch(A + (long long)(m0 + arow) * lda + k0 + 64 + acg, 0, 3);
        }

        // ---- 8 WMMAs on current buffer (fragments per ISA 7.12.2) ----
        FragU fa[2], fb[4];
        #pragma unroll
        for (int rt = 0; rt < 2; ++rt) {
            const unsigned short* ap = &As[cur][(wm * 32 + rt * 16 + lm) * 32 + hi * 8];
            fa[rt].q[0] = *(const v4u*)ap;           // K = hi*8 .. +7
            fa[rt].q[1] = *(const v4u*)(ap + 16);    // K = 16 + hi*8 .. +7
        }
        #pragma unroll
        for (int ct = 0; ct < 4; ++ct) {
            const unsigned short* bp = &Bs[cur][(wn * 64 + ct * 16 + lm) * 32 + hi * 8];
            fb[ct].q[0] = *(const v4u*)bp;
            fb[ct].q[1] = *(const v4u*)(bp + 16);
        }
        #pragma unroll
        for (int rt = 0; rt < 2; ++rt)
            #pragma unroll
            for (int ct = 0; ct < 4; ++ct)
                acc[rt][ct] = __builtin_amdgcn_wmma_f32_16x16x32_bf16(
                    false, fa[rt].v, false, fb[ct].v, (short)0, acc[rt][ct], false, false);

        if (hasNext) {                       // convert + store into alternate buffer
            storeA_l(As[cur ^ 1], ra);
            storeB_l(Bs[cur ^ 1], rb);
        }
        __syncthreads();
        cur ^= 1;
    }

    // ---- epilogue: alpha, bias, gelu, residual; guard cols for N=64 ----
    #pragma unroll
    for (int ct = 0; ct < 4; ++ct) {
        const int col = n0 + wn * 64 + ct * 16 + lm;
        if (col >= Nn) continue;
        const float bb = (flags & FLAG_BIAS) ? bias[col] : 0.0f;
        #pragma unroll
        for (int rt = 0; rt < 2; ++rt) {
            #pragma unroll
            for (int r = 0; r < 8; ++r) {
                const int row = m0 + wm * 32 + rt * 16 + hi * 8 + r;
                float t = alpha * acc[rt][ct][r] + bb;
                if (flags & FLAG_GELU) t = gelu_exact(t);
                const long long o = (long long)row * ldc + col;
                if (flags & FLAG_RES) t += R[o];
                C[o] = t;
            }
        }
    }
}

// ---------------------------------------------------------------------------
// Embedding: h[bs,d] = tok_emb[x[bs],d] + pos_emb[s,d], float4 granularity
// ---------------------------------------------------------------------------
__global__ void embed_kernel(const int* __restrict__ x, const float* __restrict__ tok,
                             const float* __restrict__ pos, float* __restrict__ h)
{
    const int i  = blockIdx.x * blockDim.x + threadIdx.x;  // float4 index
    const int d4 = i & (D_ / 4 - 1);
    const int bs = i >> 7;                                 // D_/4 == 128
    const int s  = bs & (S_ - 1);
    const int t  = x[bs];
    const float4 a = ((const float4*)tok)[(long long)t * (D_ / 4) + d4];
    const float4 p = ((const float4*)pos)[(long long)s * (D_ / 4) + d4];
    float4 o; o.x = a.x + p.x; o.y = a.y + p.y; o.z = a.z + p.z; o.w = a.w + p.w;
    ((float4*)h)[i] = o;
}

// ---------------------------------------------------------------------------
// LayerNorm over D_=512, one block per row, 256 threads x 2 elements
// ---------------------------------------------------------------------------
__global__ __launch_bounds__(256) void layernorm_kernel(
    const float* __restrict__ x, const float* __restrict__ g,
    const float* __restrict__ b, float* __restrict__ y)
{
    __shared__ float part1[8], part2[8];
    const int row = blockIdx.x;
    const int tid = threadIdx.x;
    const int wv = tid >> 5, lane = tid & 31;
    const float* xr = x + (long long)row * D_;
    const float v0 = xr[tid], v1 = xr[tid + 256];
    float s1 = v0 + v1, s2 = v0 * v0 + v1 * v1;
    #pragma unroll
    for (int m = 16; m > 0; m >>= 1) {
        s1 += __shfl_xor(s1, m, 32);
        s2 += __shfl_xor(s2, m, 32);
    }
    if (lane == 0) { part1[wv] = s1; part2[wv] = s2; }
    __syncthreads();
    float t1 = 0.f, t2 = 0.f;
    #pragma unroll
    for (int i = 0; i < 8; ++i) { t1 += part1[i]; t2 += part2[i]; }
    const float mu  = t1 * (1.0f / D_);
    const float var = t2 * (1.0f / D_) - mu * mu;
    const float rs  = rsqrtf(var + 1e-5f);
    float* yr = y + (long long)row * D_;
    yr[tid]       = (v0 - mu) * rs * g[tid]       + b[tid];
    yr[tid + 256] = (v1 - mu) * rs * g[tid + 256] + b[tid + 256];
}

// ---------------------------------------------------------------------------
// Top-K(307) + softmax over each row of 1024 scores, in place.
// Order-preserving float->uint map, 32-step radix bisection for the Kth
// largest key, then softmax over {key >= threshold}, zeros elsewhere.
// Wave32 __shfl_xor reductions + 8-entry LDS combine (2 barriers/step).
// ---------------------------------------------------------------------------
__global__ __launch_bounds__(256) void topk_softmax_kernel(float* __restrict__ sc)
{
    __shared__ int   ired[8];
    __shared__ float fred[8];
    const int tid = threadIdx.x;
    const int wv = tid >> 5, lane = tid & 31;
    float* p = sc + (long long)blockIdx.x * S_;

    float lv[4]; unsigned int key[4];
    #pragma unroll
    for (int j = 0; j < 4; ++j) {
        const float f = p[tid + j * 256];
        lv[j] = f;
        const unsigned int u = __float_as_uint(f);
        key[j] = (u & 0x80000000u) ? ~u : (u | 0x80000000u);
    }
    unsigned int thr = 0u;  // largest v with count(key >= v) >= KSEL_ == Kth largest key
    for (int bit = 31; bit >= 0; --bit) {
        const unsigned int cand = thr | (1u << bit);
        int c = 0;
        #pragma unroll
        for (int j = 0; j < 4; ++j) c += (key[j] >= cand) ? 1 : 0;
        #pragma unroll
        for (int m = 16; m > 0; m >>= 1) c += __shfl_xor(c, m, 32);
        if (lane == 0) ired[wv] = c;
        __syncthreads();
        int tot = 0;
        #pragma unroll
        for (int i = 0; i < 8; ++i) tot += ired[i];
        if (tot >= KSEL_) thr = cand;
        __syncthreads();
    }
    float mx = -3.4e38f;
    #pragma unroll
    for (int j = 0; j < 4; ++j) if (key[j] >= thr) mx = fmaxf(mx, lv[j]);
    #pragma unroll
    for (int m = 16; m > 0; m >>= 1) mx = fmaxf(mx, __shfl_xor(mx, m, 32));
    if (lane == 0) fred[wv] = mx;
    __syncthreads();
    float MX = -3.4e38f;
    #pragma unroll
    for (int i = 0; i < 8; ++i) MX = fmaxf(MX, fred[i]);
    __syncthreads();
    float e[4]; float sm = 0.0f;
    #pragma unroll
    for (int j = 0; j < 4; ++j) {
        e[j] = (key[j] >= thr) ? __expf(lv[j] - MX) : 0.0f;
        sm += e[j];
    }
    #pragma unroll
    for (int m = 16; m > 0; m >>= 1) sm += __shfl_xor(sm, m, 32);
    if (lane == 0) fred[wv] = sm;
    __syncthreads();
    float tot = 0.f;
    #pragma unroll
    for (int i = 0; i < 8; ++i) tot += fred[i];
    const float inv = 1.0f / tot;
    #pragma unroll
    for (int j = 0; j < 4; ++j) p[tid + j * 256] = e[j] * inv;
}

// ---------------------------------------------------------------------------
static void launch_gemm(hipStream_t st, const float* A, const float* Bm, float* C,
                        const float* bias, const float* R,
                        int Mrows, int Ncols, int Kd,
                        int lda, int ldb, int ldc, int Z,
                        long long sA, long long sB, long long sC, long long sR,
                        float alpha, int flags)
{
    dim3 grid((Ncols + 127) / 128, Mrows / 128, Z), blk(256);
    wmma_gemm_kernel<<<grid, blk, 0, st>>>(A, Bm, C, bias, R, Ncols, Kd,
                                           lda, ldb, ldc, sA, sB, sC, sR, alpha, flags);
}

extern "C" void kernel_launch(void* const* d_in, const int* in_sizes, int n_in,
                              void* d_out, int out_size, void* d_ws, size_t ws_size,
                              hipStream_t stream)
{
    (void)in_sizes; (void)n_in; (void)out_size; (void)ws_size;
    const int*   x    = (const int*)  d_in[0];
    const float* tok  = (const float*)d_in[1];
    const float* pos  = (const float*)d_in[2];
    const float* Wq   = (const float*)d_in[3];
    const float* bq   = (const float*)d_in[4];
    const float* Wk   = (const float*)d_in[5];
    const float* bk   = (const float*)d_in[6];
    const float* Wv   = (const float*)d_in[7];
    const float* bv   = (const float*)d_in[8];
    const float* Wo   = (const float*)d_in[9];
    const float* bo   = (const float*)d_in[10];
    const float* g1   = (const float*)d_in[11];
    const float* be1  = (const float*)d_in[12];
    const float* g2   = (const float*)d_in[13];
    const float* be2  = (const float*)d_in[14];
    const float* W1   = (const float*)d_in[15];
    const float* b1   = (const float*)d_in[16];
    const float* Wm   = (const float*)d_in[17];
    const float* bm   = (const float*)d_in[18];
    const float* W2   = (const float*)d_in[19];
    const float* b2   = (const float*)d_in[20];
    const float* Wout = (const float*)d_in[21];
    const float* bout = (const float*)d_in[22];
    float* out = (float*)d_out;

    // Workspace layout (f32), ~252 MB total
    float* ws = (float*)d_ws;
    size_t off = 0;
    float* h  = ws + off; off += (size_t)M_ * D_;
    float* y  = ws + off; off += (size_t)M_ * D_;
    float* q  = ws + off; off += (size_t)M_ * D_;
    float* kk = ws + off; off += (size_t)M_ * D_;
    float* vv = ws + off; off += (size_t)M_ * D_;
    float* o  = ws + off; off += (size_t)M_ * D_;
    float* f1 = ws + off; off += (size_t)M_ * DFF_;
    float* f2 = ws + off; off += (size_t)M_ * DFF_;
    float* sc = ws + off; off += (size_t)B_ * H_ * S_ * S_;

    embed_kernel<<<dim3(M_ * D_ / 4 / 256), dim3(256), 0, stream>>>(x, tok, pos, h);

    const float scale = 0.125f;  // 1/sqrt(DK)
    for (int l = 0; l < L_; ++l) {
        const size_t wo = (size_t)l * D_ * D_;
        layernorm_kernel<<<dim3(M_), dim3(256), 0, stream>>>(h, g1 + l * D_, be1 + l * D_, y);

        launch_gemm(stream, y, Wq + wo, q,  bq + l * D_, nullptr, M_, D_, D_, D_, D_, D_, 1,
                    0, 0, 0, 0, 1.0f, FLAG_BIAS);
        launch_gemm(stream, y, Wk + wo, kk, bk + l * D_, nullptr, M_, D_, D_, D_, D_, D_, 1,
                    0, 0, 0, 0, 1.0f, FLAG_BIAS);
        launch_gemm(stream, y, Wv + wo, vv, bv + l * D_, nullptr, M_, D_, D_, D_, D_, D_, 1,
                    0, 0, 0, 0, 1.0f, FLAG_BIAS);

        // scores[b,h] = scale * Q[b,h] @ K[b,h]^T   (grid.z = heads)
        for (int b = 0; b < B_; ++b) {
            launch_gemm(stream, q + (size_t)b * S_ * D_, kk + (size_t)b * S_ * D_,
                        sc + (size_t)b * H_ * S_ * S_, nullptr, nullptr,
                        S_, S_, DK_, D_, D_, S_, H_,
                        DK_, DK_, (long long)S_ * S_, 0, scale, FLAG_TB);
        }
        topk_softmax_kernel<<<dim3(B_ * H_ * S_), dim3(256), 0, stream>>>(sc);
        // o[b,h] = attn[b,h] @ V[b,h]
        for (int b = 0; b < B_; ++b) {
            launch_gemm(stream, sc + (size_t)b * H_ * S_ * S_, vv + (size_t)b * S_ * D_,
                        o + (size_t)b * S_ * D_, nullptr, nullptr,
                        S_, DK_, S_, S_, D_, D_, H_,
                        (long long)S_ * S_, DK_, DK_, 0, 1.0f, 0);
        }
        // h += o @ Wo + bo
        launch_gemm(stream, o, Wo + wo, h, bo + l * D_, h, M_, D_, D_, D_, D_, D_, 1,
                    0, 0, 0, 0, 1.0f, FLAG_BIAS | FLAG_RES);

        layernorm_kernel<<<dim3(M_), dim3(256), 0, stream>>>(h, g2 + l * D_, be2 + l * D_, y);
        launch_gemm(stream, y, W1 + (size_t)l * D_ * DFF_, f1, b1 + l * DFF_, nullptr,
                    M_, DFF_, D_, D_, DFF_, DFF_, 1, 0, 0, 0, 0, 1.0f, FLAG_BIAS | FLAG_GELU);
        launch_gemm(stream, f1, Wm + (size_t)l * DFF_ * DFF_, f2, bm + l * DFF_, nullptr,
                    M_, DFF_, DFF_, DFF_, DFF_, DFF_, 1, 0, 0, 0, 0, 1.0f, FLAG_BIAS | FLAG_GELU);
        launch_gemm(stream, f2, W2 + (size_t)l * DFF_ * D_, h, b2 + l * D_, h,
                    M_, D_, DFF_, DFF_, D_, D_, 1, 0, 0, 0, 0, 1.0f, FLAG_BIAS | FLAG_RES);
    }
    // logits = h @ Wout + bout
    launch_gemm(stream, h, Wout, out, bout, nullptr,
                M_, V_, D_, D_, V_, V_, 1, 0, 0, 0, 0, 1.0f, FLAG_BIAS);
}
